// QuantizedMlp_63136019251234
// MI455X (gfx1250) — 8-bit matrix operand order repaired and checked by reading, compiled, not run on hardware
//
#include <hip/hip_runtime.h>


typedef __attribute__((ext_vector_type(8))) int v8i;
typedef __attribute__((ext_vector_type(4))) int v4i;
typedef __attribute__((ext_vector_type(2))) int v2i;

static constexpr int IN_F  = 1024;
static constexpr int HID_F = 4096;
static constexpr int OUT_F = 1024;
static constexpr int MROWS = 4 * 2048;

static constexpr int TPITCH      = 80;
static constexpr int TILE_BYTES  = 128 * TPITCH;
static constexpr int STAGE_BYTES = 2 * TILE_BYTES;

__device__ __forceinline__ void async_copy_b128(unsigned lds_addr,
                                                unsigned long long gaddr) {
    asm volatile("global_load_async_to_lds_b128 %0, %1, off"
                 :: "v"(lds_addr), "v"(gaddr)
                 : "memory");
}

template <int N>
__device__ __forceinline__ void wait_asynccnt() {
    asm volatile("s_wait_asynccnt %0" :: "n"(N));
}

__device__ __forceinline__ int quant_i8(float x, float scale) {
    float r = rintf(x * scale);
    r = fminf(fmaxf(r, -128.0f), 127.0f);
    return (int)r;
}

__global__ __launch_bounds__(256)
void quant4_kernel(const float* __restrict__ in, signed char* __restrict__ outp,
                   int n4, float scale) {
    int i = blockIdx.x * blockDim.x + threadIdx.x;
    if (i >= n4) return;
    float4 v = ((const float4*)in)[i];
    int q0 = quant_i8(v.x, scale) & 255;
    int q1 = quant_i8(v.y, scale) & 255;
    int q2 = quant_i8(v.z, scale) & 255;
    int q3 = quant_i8(v.w, scale) & 255;
    ((int*)outp)[i] = q0 | (q1 << 8) | (q2 << 16) | (q3 << 24);
}

template <bool RELUQ, int K>
__global__ __launch_bounds__(256)
void gemm_i8_wmma(const signed char* __restrict__ A,
                  const signed char* __restrict__ W,
                  const float* __restrict__ bias,
                  void* __restrict__ outp,
                  int N) {
    __shared__ char smem[2 * STAGE_BYTES];

    const int tid   = threadIdx.x;
    const int lane  = tid & 31;
    const int wave  = tid >> 5;
    const int waveM = wave & 1;
    const int waveN = wave >> 1;
    const int g     = lane >> 4;
    const int l16   = lane & 15;

    const int mBlock = blockIdx.x * 128;
    const int nBlock = blockIdx.y * 128;

    const unsigned lds_base =
        (unsigned)(uintptr_t)(__attribute__((address_space(3))) char*)smem;

    const int crow = tid >> 2;
    const int csub = (tid & 3) * 16;

    const unsigned long long gA0 =
        (unsigned long long)(A + (size_t)(mBlock + crow) * K + csub);
    const unsigned long long gB0 =
        (unsigned long long)(W + (size_t)(nBlock + crow) * K + csub);
    const unsigned long long gRowSkip = (unsigned long long)64 * K;

    const unsigned lA0 = lds_base + crow * TPITCH + csub;
    const unsigned lB0 = lA0 + TILE_BYTES;

    v8i c[4][2] = {};

    auto issue_stage = [&](int k, int buf) {
        const unsigned lb = buf * STAGE_BYTES;
        async_copy_b128(lA0 + lb, gA0 + k);
        async_copy_b128(lA0 + lb + 64u * TPITCH, gA0 + gRowSkip + k);
        async_copy_b128(lB0 + lb, gB0 + k);
        async_copy_b128(lB0 + lb + 64u * TPITCH, gB0 + gRowSkip + k);
    };

    issue_stage(0, 0);

    int buf = 0;
    for (int k = 0; k < K; k += 64) {
        if (k + 64 < K) {
            issue_stage(k + 64, buf ^ 1);
            wait_asynccnt<4>();
        } else {
            wait_asynccnt<0>();
        }
        __syncthreads();

        const char* aTile = smem + buf * STAGE_BYTES;
        const char* bTile = aTile + TILE_BYTES;

        v8i bf[2];
#pragma unroll
        for (int j = 0; j < 2; ++j) {
            const char* p = bTile + (waveN * 32 + j * 16 + l16) * TPITCH + g * 16;
            v4i lo = *(const v4i*)(p);
            v4i hi = *(const v4i*)(p + 32);
            bf[j][0] = lo[0]; bf[j][1] = lo[1]; bf[j][2] = lo[2]; bf[j][3] = lo[3];
            bf[j][4] = hi[0]; bf[j][5] = hi[1]; bf[j][6] = hi[2]; bf[j][7] = hi[3];
        }
        v8i af[4];
#pragma unroll
        for (int i = 0; i < 4; ++i) {
            const char* p = aTile + (waveM * 64 + i * 16 + l16) * TPITCH + g * 16;
            v2i t0 = *(const v2i*)(p);
            v2i t1 = *(const v2i*)(p + 8);
            v2i t2 = *(const v2i*)(p + 32);
            v2i t3 = *(const v2i*)(p + 40);
            af[i][0] = t0[0]; af[i][1] = t0[1];
            af[i][2] = t1[0]; af[i][3] = t1[1];
            af[i][4] = t2[0]; af[i][5] = t2[1];
            af[i][6] = t3[0]; af[i][7] = t3[1];
        }

#pragma unroll
        for (int i = 0; i < 4; ++i)
#pragma unroll
            for (int j = 0; j < 2; ++j)
                c[i][j] = __builtin_amdgcn_wmma_i32_16x16x64_iu8(
                     true, af[i],  true, bf[j],
                    c[i][j],  false,  false);

        __syncthreads();
        buf ^= 1;
    }

    const float inv_prod = 1.0f / 1024.0f;
#pragma unroll
    for (int j = 0; j < 2; ++j) {
        const int n = nBlock + waveN * 32 + j * 16 + l16;
        const float qb =
            fminf(fmaxf(rintf(bias[n] * 64.0f), -128.0f), 127.0f) * (1.0f / 64.0f);
#pragma unroll
        for (int i = 0; i < 4; ++i) {
            const int mrow = mBlock + waveM * 64 + i * 16 + g * 8;
#pragma unroll
            for (int r = 0; r < 8; ++r) {
                float h = (float)c[i][j][r] * inv_prod + qb;
                if (RELUQ) {
                    int q = quant_i8(h, 16.0f);
                    q = q > 0 ? q : 0;
                    ((signed char*)outp)[(size_t)(mrow + r) * N + n] = (signed char)q;
                } else {
                    ((float*)outp)[(size_t)(mrow + r) * N + n] = h;
                }
            }
        }
    }
}

extern "C" void kernel_launch(void* const* d_in, const int* in_sizes, int n_in,
                              void* d_out, int out_size, void* d_ws, size_t ws_size,
                              hipStream_t stream) {
    const float* x  = (const float*)d_in[0];
    const float* w1 = (const float*)d_in[1];
    const float* b1 = (const float*)d_in[2];
    const float* w2 = (const float*)d_in[3];
    const float* b2 = (const float*)d_in[4];
    float* out = (float*)d_out;

    signed char* xq  = (signed char*)d_ws;
    signed char* w1q = xq + (size_t)MROWS * IN_F;
    signed char* w2q = w1q + (size_t)HID_F * IN_F;
    signed char* hq  = w2q + (size_t)OUT_F * HID_F;

    {
        int n4 = (MROWS * IN_F) / 4;
        quant4_kernel<<<(n4 + 255) / 256, 256, 0, stream>>>(x, xq, n4, 16.0f);
    }
    {
        int n4 = (HID_F * IN_F) / 4;
        quant4_kernel<<<(n4 + 255) / 256, 256, 0, stream>>>(w1, w1q, n4, 64.0f);
    }
    {
        int n4 = (OUT_F * HID_F) / 4;
        quant4_kernel<<<(n4 + 255) / 256, 256, 0, stream>>>(w2, w2q, n4, 64.0f);
    }

    {
        dim3 grid(MROWS / 128, HID_F / 128);
        gemm_i8_wmma<true, IN_F><<<grid, 256, 0, stream>>>(xq, w1q, b1, hq, HID_F);
    }

    {
        dim3 grid(MROWS / 128, OUT_F / 128);
        gemm_i8_wmma<false, HID_F><<<grid, 256, 0, stream>>>(hq, w2q, b2, out, OUT_F);
    }
}
